// Model4_Hybrid_63264868270721
// MI455X (gfx1250) — matrix operand order repaired and checked by reading, compiled, not run on hardware
//
#include <hip/hip_runtime.h>
#include <math.h>

#define NNODES 50000
#define NEDGES 400000
#define NGRAPH 64

typedef __attribute__((ext_vector_type(16))) _Float16 v16h;
typedef __attribute__((ext_vector_type(8)))  _Float16 v8h;
typedef __attribute__((ext_vector_type(8)))  float    v8f;

// ---------- helpers ----------
__device__ __forceinline__ unsigned fenc(float f) {
    unsigned u = __float_as_uint(f);
    return (u & 0x80000000u) ? ~u : (u | 0x80000000u);
}
__device__ __forceinline__ float fdec(unsigned u) {
    return __uint_as_float((u & 0x80000000u) ? (u & 0x7FFFFFFFu) : ~u);
}
__device__ __forceinline__ float geluf(float x) {
    return 0.5f * x * (1.0f + erff(x * 0.70710678118654752f));
}

// ---------- layout prep: f32 [M,K] -> f16 [M,Kp] zero-padded ----------
__global__ void a_to_f16(const float* __restrict__ src, _Float16* __restrict__ dst,
                         int M, int K, int Kp)
{
    long long i = (long long)blockIdx.x * blockDim.x + threadIdx.x;
    long long total = (long long)M * Kp;
    if (i >= total) return;
    int m = (int)(i / Kp), k = (int)(i % Kp);
    dst[i] = (k < K) ? (_Float16)src[(size_t)m * K + k] : (_Float16)0.0f;
}
// ---------- weight prep: f32 [K,N] -> f16 transposed [N,Kp] zero-padded ----------
__global__ void w_to_f16t(const float* __restrict__ W, _Float16* __restrict__ Wt,
                          int K, int N, int Kp)
{
    int i = blockIdx.x * blockDim.x + threadIdx.x;
    if (i >= N * Kp) return;
    int n = i / Kp, k = i % Kp;
    Wt[i] = (k < K) ? (_Float16)W[(size_t)k * N + n] : (_Float16)0.0f;
}

// ---------- WMMA GEMM: Out[M,N] = A16[M,Kp] * Wt16[N,Kp]^T (+bias) ----------
// Branch-free: Kp % 32 == 0, M % 16 == 0, N % 16 == 0. One wave = one 16x16 tile.
__global__ void __launch_bounds__(256) wmma_gemm(
    const _Float16* __restrict__ A, const _Float16* __restrict__ Wt,
    const float* __restrict__ bias, float* __restrict__ Out, int Kp, int N)
{
    const int lane   = (int)threadIdx.x & 31;
    const int wave   = (int)threadIdx.x >> 5;
    const int tile_m = blockIdx.x << 4;
    const int n0     = (blockIdx.y << 7) + (wave << 4);
    if (n0 >= N) return;                       // no barriers in kernel -> safe
    const int row = tile_m + (lane & 15);
    const int ak  = (lane >> 4) << 3;          // A K-offset 0/8 (ISA 16-bit A layout)
    const int col = n0 + (lane & 15);
    const int bk  = (lane >> 4) << 3;          // B K-offset 0/8 (matches A layout)
    const _Float16* ap = A + (size_t)row * Kp + ak;
    const _Float16* bp = Wt + (size_t)col * Kp + bk;
    v8f c = {};
    #pragma unroll 2
    for (int kc = 0; kc < Kp; kc += 32) {
        v8h a0 = *(const v8h*)(ap + kc);       // K = kc+ak   .. +7  (contiguous)
        v8h a1 = *(const v8h*)(ap + kc + 16);  // K = kc+16+ak.. +7  (contiguous)
        v16h a;
        #pragma unroll
        for (int t = 0; t < 8; ++t) { a[t] = a0[t]; a[t + 8] = a1[t]; }
        v8h b0 = *(const v8h*)(bp + kc);
        v8h b1 = *(const v8h*)(bp + kc + 16);
        v16h b;
        #pragma unroll
        for (int t = 0; t < 8; ++t) { b[t] = b0[t]; b[t + 8] = b1[t]; }
        c = __builtin_amdgcn_wmma_f32_16x16x32_f16(false, a, false, b, (short)0, c,
                                                   false, false);
    }
    const float bv = bias ? bias[col] : 0.0f;
    #pragma unroll
    for (int r = 0; r < 8; ++r) {
        int orow = tile_m + r + ((lane >> 4) << 3);
        Out[(size_t)orow * N + col] = c[r] + bv;
    }
}

// ---------- self-loop edge_attr mean ----------
__global__ void ea_accum(const float* __restrict__ ea, const int* __restrict__ dst,
                         float* __restrict__ easum, float* __restrict__ cnt, int E)
{
    int e = blockIdx.x * blockDim.x + threadIdx.x;
    if (e >= E) return;
    int d = dst[e];
    #pragma unroll
    for (int j = 0; j < 7; ++j) atomicAdd(&easum[(size_t)d * 7 + j], ea[(size_t)e * 7 + j]);
    atomicAdd(&cnt[d], 1.0f);
}
__global__ void ea_mean(float* __restrict__ easum, const float* __restrict__ cnt, int n)
{
    int i = blockIdx.x * blockDim.x + threadIdx.x;
    if (i >= n) return;
    float c = fmaxf(cnt[i], 1.0f);
    #pragma unroll
    for (int j = 0; j < 7; ++j) easum[(size_t)i * 7 + j] /= c;
}

// ---------- GATv2 pass 1: per-(edge,head) attention score + segment max ----------
__global__ void gat_score(const float* __restrict__ xl, const float* __restrict__ xr,
                          const float* __restrict__ ea, const float* __restrict__ ea_loop,
                          const int* __restrict__ src, const int* __restrict__ dst,
                          const float* __restrict__ We, const float* __restrict__ att,
                          float* __restrict__ score, unsigned* __restrict__ smax,
                          int E, int Nn, int H)
{
    __shared__ float sWe[7 * 256];
    __shared__ float sAtt[256];
    const int F = H << 6;
    for (int j = (int)threadIdx.x; j < 7 * F; j += blockDim.x) sWe[j] = We[j];
    for (int j = (int)threadIdx.x; j < F; j += blockDim.x) sAtt[j] = att[j];
    __syncthreads();
    long long i = (long long)blockIdx.x * blockDim.x + threadIdx.x;
    long long total = (long long)(E + Nn) * H;
    if (i >= total) return;
    int e = (int)(i / H), h = (int)(i % H);
    int s, d; const float* eap;
    if (e < E) { s = src[e]; d = dst[e]; eap = ea + (size_t)e * 7; }
    else       { s = d = e - E; eap = ea_loop + (size_t)s * 7; }
    float eav[7];
    #pragma unroll
    for (int k = 0; k < 7; ++k) eav[k] = eap[k];
    const float4* xls4 = (const float4*)(xl + (size_t)s * F + (h << 6));
    const float4* xrd4 = (const float4*)(xr + (size_t)d * F + (h << 6));
    float sc = 0.0f;
    for (int q = 0; q < 16; ++q) {
        float4 va = xls4[q];
        float4 vb = xrd4[q];
        float mm[4] = {va.x + vb.x, va.y + vb.y, va.z + vb.z, va.w + vb.w};
        #pragma unroll
        for (int j = 0; j < 4; ++j) {
            int f = (h << 6) + (q << 2) + j;
            float ee = 0.0f;
            #pragma unroll
            for (int k = 0; k < 7; ++k) ee = fmaf(eav[k], sWe[k * F + f], ee);
            float m = mm[j] + ee;
            m = (m > 0.0f) ? m : 0.2f * m;       // leaky_relu(0.2)
            sc = fmaf(m, sAtt[f], sc);
        }
    }
    score[i] = sc;
    atomicMax(&smax[(size_t)d * H + h], fenc(sc));
}

// ---------- GATv2 pass 2: exp(score - max), segment sum denominator ----------
__global__ void gat_expdenom(float* __restrict__ score, const unsigned* __restrict__ smax,
                             float* __restrict__ denom, const int* __restrict__ dst,
                             int E, int Nn, int H)
{
    long long i = (long long)blockIdx.x * blockDim.x + threadIdx.x;
    long long total = (long long)(E + Nn) * H;
    if (i >= total) return;
    int e = (int)(i / H), h = (int)(i % H);
    int d = (e < E) ? dst[e] : (e - E);
    float m = fdec(smax[(size_t)d * H + h]);
    if (!(m > -1.0e38f && m < 1.0e38f)) m = 0.0f;   // mirror where(isfinite, m, 0)
    float ev = __expf(score[i] - m);
    score[i] = ev;
    atomicAdd(&denom[(size_t)d * H + h], ev);
}

// ---------- GATv2 pass 3: out[d] += xl[s] * alpha ----------
__global__ void gat_accum(const float* __restrict__ xl, const float* __restrict__ escore,
                          const float* __restrict__ denom, const int* __restrict__ src,
                          const int* __restrict__ dst, float* __restrict__ acc,
                          int E, int Nn, int H)
{
    const int F = H << 6;
    long long i = (long long)blockIdx.x * blockDim.x + threadIdx.x;
    long long total = (long long)(E + Nn) * F;
    if (i >= total) return;
    int e = (int)(i / F), f = (int)(i % F), h = f >> 6;
    int s, d;
    if (e < E) { s = src[e]; d = dst[e]; } else { s = d = e - E; }
    float alpha = escore[(size_t)e * H + h] / (denom[(size_t)d * H + h] + 1e-16f);
    atomicAdd(&acc[(size_t)d * F + f], xl[(size_t)s * F + f] * alpha);
}

// ---------- epilogues (activation outputs that feed GEMMs are written as f16) ----------
__global__ void bias_gelu_to_f16(const float* __restrict__ a, const float* __restrict__ b,
                                 _Float16* __restrict__ out16, int F, long long total)
{
    long long i = (long long)blockIdx.x * blockDim.x + threadIdx.x;
    if (i >= total) return;
    out16[i] = (_Float16)geluf(a[i] + b[(int)(i % F)]);
}
__global__ void headmean_bias_gelu(const float* __restrict__ acc, const float* __restrict__ b,
                                   float* __restrict__ out, int n)
{
    int i = blockIdx.x * blockDim.x + threadIdx.x;
    if (i >= n) return;
    int node = i >> 6, cch = i & 63;
    float v = 0.5f * (acc[(size_t)node * 128 + cch] + acc[(size_t)node * 128 + 64 + cch]) + b[cch];
    out[i] = geluf(v);
}
__global__ void bias_relu_to_f16(const float* __restrict__ a, const float* __restrict__ b,
                                 _Float16* __restrict__ out16, int F, long long total)
{
    long long i = (long long)blockIdx.x * blockDim.x + threadIdx.x;
    if (i >= total) return;
    out16[i] = (_Float16)fmaxf(a[i] + b[(int)(i % F)], 0.0f);
}
__global__ void bias_relu_ip(float* __restrict__ a, const float* __restrict__ b,
                             int F, long long total)
{
    long long i = (long long)blockIdx.x * blockDim.x + threadIdx.x;
    if (i >= total) return;
    a[i] = fmaxf(a[i] + b[(int)(i % F)], 0.0f);
}

// ---------- GCN ----------
__global__ void deg_accum(const int* __restrict__ dst, float* __restrict__ deg, int E)
{
    int e = blockIdx.x * blockDim.x + threadIdx.x;
    if (e >= E) return;
    atomicAdd(&deg[dst[e]], 1.0f);
}
__global__ void deg_to_dinv(float* __restrict__ deg, int n)
{
    int i = blockIdx.x * blockDim.x + threadIdx.x;
    if (i >= n) return;
    deg[i] = rsqrtf(deg[i] + 1.0f);              // +1 self loop; always >= 1
}
__global__ void gcn_prop(const float* __restrict__ h, const float* __restrict__ dinv,
                         const int* __restrict__ src, const int* __restrict__ dst,
                         float* __restrict__ acc, int E, int Nn)
{
    long long i = (long long)blockIdx.x * blockDim.x + threadIdx.x;
    long long total = (long long)(E + Nn) * 64;
    if (i >= total) return;
    int e = (int)(i >> 6), cch = (int)(i & 63);
    int s, d;
    if (e < E) { s = src[e]; d = dst[e]; } else { s = d = e - E; }
    atomicAdd(&acc[(size_t)d * 64 + cch], h[(size_t)s * 64 + cch] * dinv[s] * dinv[d]);
}

// ---------- pooling ----------
__global__ void batch_count(const int* __restrict__ batch, float* __restrict__ cntg, int n)
{
    int i = blockIdx.x * blockDim.x + threadIdx.x;
    if (i >= n) return;
    atomicAdd(&cntg[batch[i]], 1.0f);
}
__global__ void pool_accum(const float* __restrict__ h, const int* __restrict__ batch,
                           float* __restrict__ psum, unsigned* __restrict__ pmax, int n)
{
    int i = blockIdx.x * blockDim.x + threadIdx.x;
    if (i >= n) return;
    int node = i >> 6, c = i & 63;
    int g = batch[node];
    float v = h[i];
    atomicAdd(&psum[g * 64 + c], v);
    atomicMax(&pmax[g * 64 + c], fenc(v));
}
__global__ void pool_final(const float* __restrict__ psum_h, const unsigned* __restrict__ pmax_h,
                           const float* __restrict__ psum_c, const unsigned* __restrict__ pmax_c,
                           const float* __restrict__ cntg, float* __restrict__ pooled)
{
    int i = blockIdx.x * blockDim.x + threadIdx.x;
    if (i >= NGRAPH * 384) return;
    int g = i / 384, j = i % 384;
    float cn = fmaxf(cntg[g], 1.0f);
    float v;
    if (j < 64)        v = psum_h[g * 64 + j] / cn;
    else if (j < 128)  v = fdec(pmax_h[g * 64 + j - 64]);
    else if (j < 192)  v = psum_h[g * 64 + j - 128];
    else if (j < 256)  v = psum_c[g * 64 + j - 192] / cn;
    else if (j < 320)  v = fdec(pmax_c[g * 64 + j - 256]);
    else               v = psum_c[g * 64 + j - 320];
    pooled[i] = v;
}

// ---------- MLP head: one block per graph, LDS layernorm chain ----------
__global__ void __launch_bounds__(384) mlp_head(
    const float* __restrict__ pooled,
    const float* W0, const float* b0, const float* g0, const float* be0,
    const float* W1, const float* b1, const float* g1, const float* be1,
    const float* W2, const float* b2, const float* g2, const float* be2,
    const float* W3, const float* b3, const float* g3, const float* be3,
    const float* W4, const float* b4,
    float* __restrict__ out)
{
    __shared__ float z[384];
    __shared__ float red[384];
    __shared__ float s_mu, s_rs;
    const int tid = threadIdx.x;
    const int g = blockIdx.x;
    const float* Ws[5]  = {W0, W1, W2, W3, W4};
    const float* bs[5]  = {b0, b1, b2, b3, b4};
    const float* gs[4]  = {g0, g1, g2, g3};
    const float* bes[4] = {be0, be1, be2, be3};
    const int dims[6] = {384, 320, 256, 128, 64, 9};
    z[tid] = pooled[g * 384 + tid];
    __syncthreads();
    for (int L = 0; L < 5; ++L) {
        int din = dims[L], dout = dims[L + 1];
        float a = 0.0f;
        if (tid < dout) {
            const float* W = Ws[L];
            for (int k = 0; k < din; ++k) a = fmaf(z[k], W[k * dout + tid], a);
            a += bs[L][tid];
        }
        if (L == 4) {
            if (tid < dout) out[g * 9 + tid] = a;
            return;
        }
        a = (tid < dout) ? geluf(a) : 0.0f;
        red[tid] = a;
        __syncthreads();
        if (tid == 0) { float s = 0.0f; for (int k = 0; k < dout; ++k) s += red[k]; s_mu = s / dout; }
        __syncthreads();
        float dv = (tid < dout) ? (a - s_mu) : 0.0f;
        red[tid] = dv * dv;
        __syncthreads();
        if (tid == 0) { float s = 0.0f; for (int k = 0; k < dout; ++k) s += red[k]; s_rs = rsqrtf(s / dout + 1e-5f); }
        __syncthreads();
        z[tid] = (tid < dout) ? (dv * s_rs * gs[L][tid] + bes[L][tid]) : 0.0f;
        __syncthreads();
    }
}

// ---------- host ----------
extern "C" void kernel_launch(void* const* d_in, const int* in_sizes, int n_in,
                              void* d_out, int out_size, void* d_ws, size_t ws_size,
                              hipStream_t stream)
{
    (void)in_sizes; (void)n_in; (void)out_size;
    const float* x      = (const float*)d_in[0];
    const float* eattr  = (const float*)d_in[1];
    const float* g1_Wl  = (const float*)d_in[2];
    const float* g1_bl  = (const float*)d_in[3];
    const float* g1_Wr  = (const float*)d_in[4];
    const float* g1_br  = (const float*)d_in[5];
    const float* g1_We  = (const float*)d_in[6];
    const float* g1_att = (const float*)d_in[7];
    const float* g1_b   = (const float*)d_in[8];
    const float* g2_Wl  = (const float*)d_in[9];
    const float* g2_bl  = (const float*)d_in[10];
    const float* g2_Wr  = (const float*)d_in[11];
    const float* g2_br  = (const float*)d_in[12];
    const float* g2_We  = (const float*)d_in[13];
    const float* g2_att = (const float*)d_in[14];
    const float* g2_b   = (const float*)d_in[15];
    const float* c1_W   = (const float*)d_in[16];
    const float* c1_b   = (const float*)d_in[17];
    const float* c2_W   = (const float*)d_in[18];
    const float* c2_b   = (const float*)d_in[19];
    const float *hW[5], *hb[5], *hg[4], *hbe[4];
    int idx = 20;
    for (int i = 0; i < 5; ++i) {
        hW[i] = (const float*)d_in[idx++];
        hb[i] = (const float*)d_in[idx++];
        if (i < 4) { hg[i] = (const float*)d_in[idx++]; hbe[i] = (const float*)d_in[idx++]; }
    }
    const int* eidx  = (const int*)d_in[idx++];
    const int* batch = (const int*)d_in[idx++];
    const int* src = eidx;
    const int* dst = eidx + NEDGES;

    // workspace bump allocator (256B aligned), sizes in bytes
    char* w = (char*)d_ws;
    size_t off = 0;
    auto allocB = [&](size_t bytes) -> void* {
        void* p = (void*)(w + off);
        off += ((bytes + 255) & ~(size_t)255);
        return p;
    };
    auto allocF = [&](size_t n) -> float*     { return (float*)allocB(n * 4); };
    auto allocU = [&](size_t n) -> unsigned*  { return (unsigned*)allocB(n * 4); };
    auto allocH = [&](size_t n) -> _Float16*  { return (_Float16*)allocB(n * 2); };

    float* xl1    = allocF((size_t)NNODES * 256);
    float* xr1    = allocF((size_t)NNODES * 256);
    float* acc1   = allocF((size_t)NNODES * 256);
    float* xl2    = allocF((size_t)NNODES * 128);
    float* xr2    = allocF((size_t)NNODES * 128);
    float* acc2   = allocF((size_t)NNODES * 128);
    float* h2     = allocF((size_t)NNODES * 64);
    float* ealoop = allocF((size_t)NNODES * 7);
    float* cntn   = allocF(NNODES);
    float* score1 = allocF((size_t)(NEDGES + NNODES) * 4);
    unsigned* smax1 = allocU((size_t)NNODES * 4);
    float* den1   = allocF((size_t)NNODES * 4);
    float* score2 = allocF((size_t)(NEDGES + NNODES) * 2);
    unsigned* smax2 = allocU((size_t)NNODES * 2);
    float* den2   = allocF((size_t)NNODES * 2);
    float* deg    = allocF(NNODES);                 // becomes dinv in place
    float* c1a    = allocF((size_t)NNODES * 64);
    float* c2h    = allocF((size_t)NNODES * 64);
    float* c2a    = allocF((size_t)NNODES * 64);
    float* c1h    = allocF((size_t)NNODES * 64);
    float* psumh  = allocF(NGRAPH * 64);
    unsigned* pmaxh = allocU(NGRAPH * 64);
    float* psumc  = allocF(NGRAPH * 64);
    unsigned* pmaxc = allocU(NGRAPH * 64);
    float* cntg   = allocF(NGRAPH);
    float* pooled = allocF(NGRAPH * 384);
    // f16 staging for WMMA operands
    _Float16* x16    = allocH((size_t)NNODES * 32);   // x padded K 14->32
    _Float16* h1_16  = allocH((size_t)NNODES * 256);  // gelu(GAT1) output
    _Float16* c1_16  = allocH((size_t)NNODES * 64);   // relu(GCN1) output
    _Float16* Wt_g1l = allocH(256 * 32);
    _Float16* Wt_g1r = allocH(256 * 32);
    _Float16* Wt_g2l = allocH(128 * 256);
    _Float16* Wt_g2r = allocH(128 * 256);
    _Float16* Wt_c1  = allocH(64 * 32);
    _Float16* Wt_c2  = allocH(64 * 64);
    size_t used = off;
    if (used > ws_size) return;

    hipMemsetAsync(d_ws, 0, used, stream);

    const int ET = NEDGES + NNODES;
    // ---- operand prep ----
    a_to_f16<<<((NNODES * 32) + 255) / 256, 256, 0, stream>>>(x, x16, NNODES, 14, 32);
    w_to_f16t<<<(256 * 32 + 255) / 256, 256, 0, stream>>>(g1_Wl, Wt_g1l, 14, 256, 32);
    w_to_f16t<<<(256 * 32 + 255) / 256, 256, 0, stream>>>(g1_Wr, Wt_g1r, 14, 256, 32);
    w_to_f16t<<<(128 * 256 + 255) / 256, 256, 0, stream>>>(g2_Wl, Wt_g2l, 256, 128, 256);
    w_to_f16t<<<(128 * 256 + 255) / 256, 256, 0, stream>>>(g2_Wr, Wt_g2r, 256, 128, 256);
    w_to_f16t<<<(64 * 32 + 255) / 256, 256, 0, stream>>>(c1_W, Wt_c1, 14, 64, 32);
    w_to_f16t<<<(64 * 64 + 255) / 256, 256, 0, stream>>>(c2_W, Wt_c2, 64, 64, 64);

    // ---- GAT1 node transforms (WMMA) ----
    wmma_gemm<<<dim3(NNODES / 16, 2), 256, 0, stream>>>(x16, Wt_g1l, g1_bl, xl1, 32, 256);
    wmma_gemm<<<dim3(NNODES / 16, 2), 256, 0, stream>>>(x16, Wt_g1r, g1_br, xr1, 32, 256);
    // ---- self-loop edge_attr mean ----
    ea_accum<<<(NEDGES + 255) / 256, 256, 0, stream>>>(eattr, dst, ealoop, cntn, NEDGES);
    ea_mean<<<(NNODES + 255) / 256, 256, 0, stream>>>(ealoop, cntn, NNODES);
    // ---- GAT1 edge passes ----
    long long t1 = (long long)ET * 4;
    gat_score<<<(unsigned)((t1 + 255) / 256), 256, 0, stream>>>(xl1, xr1, eattr, ealoop, src, dst,
                                                                g1_We, g1_att, score1, smax1,
                                                                NEDGES, NNODES, 4);
    gat_expdenom<<<(unsigned)((t1 + 255) / 256), 256, 0, stream>>>(score1, smax1, den1, dst,
                                                                   NEDGES, NNODES, 4);
    long long t1f = (long long)ET * 256;
    gat_accum<<<(unsigned)((t1f + 255) / 256), 256, 0, stream>>>(xl1, score1, den1, src, dst,
                                                                 acc1, NEDGES, NNODES, 4);
    bias_gelu_to_f16<<<(unsigned)(((long long)NNODES * 256 + 255) / 256), 256, 0, stream>>>(
        acc1, g1_b, h1_16, 256, (long long)NNODES * 256);
    // ---- GAT2 node transforms (WMMA) ----
    wmma_gemm<<<dim3(NNODES / 16, 1), 256, 0, stream>>>(h1_16, Wt_g2l, g2_bl, xl2, 256, 128);
    wmma_gemm<<<dim3(NNODES / 16, 1), 256, 0, stream>>>(h1_16, Wt_g2r, g2_br, xr2, 256, 128);
    // ---- GAT2 edge passes ----
    long long t2 = (long long)ET * 2;
    gat_score<<<(unsigned)((t2 + 255) / 256), 256, 0, stream>>>(xl2, xr2, eattr, ealoop, src, dst,
                                                                g2_We, g2_att, score2, smax2,
                                                                NEDGES, NNODES, 2);
    gat_expdenom<<<(unsigned)((t2 + 255) / 256), 256, 0, stream>>>(score2, smax2, den2, dst,
                                                                   NEDGES, NNODES, 2);
    long long t2f = (long long)ET * 128;
    gat_accum<<<(unsigned)((t2f + 255) / 256), 256, 0, stream>>>(xl2, score2, den2, src, dst,
                                                                 acc2, NEDGES, NNODES, 2);
    headmean_bias_gelu<<<(NNODES * 64 + 255) / 256, 256, 0, stream>>>(acc2, g2_b, h2, NNODES * 64);
    // ---- GCN path ----
    deg_accum<<<(NEDGES + 255) / 256, 256, 0, stream>>>(dst, deg, NEDGES);
    deg_to_dinv<<<(NNODES + 255) / 256, 256, 0, stream>>>(deg, NNODES);
    wmma_gemm<<<dim3(NNODES / 16, 1), 256, 0, stream>>>(x16, Wt_c1, nullptr, c1h, 32, 64);
    long long tg = (long long)ET * 64;
    gcn_prop<<<(unsigned)((tg + 255) / 256), 256, 0, stream>>>(c1h, deg, src, dst, c1a, NEDGES, NNODES);
    bias_relu_to_f16<<<(unsigned)(((long long)NNODES * 64 + 255) / 256), 256, 0, stream>>>(
        c1a, c1_b, c1_16, 64, (long long)NNODES * 64);
    wmma_gemm<<<dim3(NNODES / 16, 1), 256, 0, stream>>>(c1_16, Wt_c2, nullptr, c2h, 64, 64);
    gcn_prop<<<(unsigned)((tg + 255) / 256), 256, 0, stream>>>(c2h, deg, src, dst, c2a, NEDGES, NNODES);
    bias_relu_ip<<<(unsigned)(((long long)NNODES * 64 + 255) / 256), 256, 0, stream>>>(
        c2a, c2_b, 64, (long long)NNODES * 64);
    // ---- pooling ----
    batch_count<<<(NNODES + 255) / 256, 256, 0, stream>>>(batch, cntg, NNODES);
    pool_accum<<<(NNODES * 64 + 255) / 256, 256, 0, stream>>>(h2, batch, psumh, pmaxh, NNODES * 64);
    pool_accum<<<(NNODES * 64 + 255) / 256, 256, 0, stream>>>(c2a, batch, psumc, pmaxc, NNODES * 64);
    pool_final<<<(NGRAPH * 384 + 255) / 256, 256, 0, stream>>>(psumh, pmaxh, psumc, pmaxc, cntg, pooled);
    // ---- MLP head ----
    mlp_head<<<NGRAPH, 384, 0, stream>>>(pooled,
        hW[0], hb[0], hg[0], hbe[0],
        hW[1], hb[1], hg[1], hbe[1],
        hW[2], hb[2], hg[2], hbe[2],
        hW[3], hb[3], hg[3], hbe[3],
        hW[4], hb[4], (float*)d_out);
}
